// self_attention_head_18872086298680
// MI455X (gfx1250) — matrix operand order repaired and checked by reading, not compiled, not run on hardware
//
#include <hip/hip_runtime.h>
#include <math.h>

#define BATCH 8
#define TSEQ  2048
#define CDIM  1024
#define HDIM  64

typedef __attribute__((ext_vector_type(16))) __bf16 v16bf;
typedef __attribute__((ext_vector_type(8)))  __bf16 v8bf;
typedef __attribute__((ext_vector_type(8)))  float  v8f;

static __device__ __forceinline__ v8f wmma_bf16(v16bf a, v16bf b, v8f c) {
  // D = A(16x32 bf16) * B(32x16 bf16) + C(16x16 f32)
  return __builtin_amdgcn_wmma_f32_16x16x32_bf16(
      /*neg_a=*/false, a, /*neg_b=*/false, b,
      /*c_mod=*/(short)0, c, /*reuse_a=*/false, /*reuse_b=*/false);
}

// Reduce across the 16 lanes of each wave32 half (the N dimension of C/D tiles).
static __device__ __forceinline__ float hmax16(float x) {
  x = fmaxf(x, __shfl_xor(x, 1, 32));
  x = fmaxf(x, __shfl_xor(x, 2, 32));
  x = fmaxf(x, __shfl_xor(x, 4, 32));
  x = fmaxf(x, __shfl_xor(x, 8, 32));
  return x;
}
static __device__ __forceinline__ float hsum16(float x) {
  x += __shfl_xor(x, 1, 32);
  x += __shfl_xor(x, 2, 32);
  x += __shfl_xor(x, 4, 32);
  x += __shfl_xor(x, 8, 32);
  return x;
}

// ---------------------------------------------------------------------------
// Kernel 0: swizzle Wq/Wk/Wv (fp32, C x H) into bf16 B-matrix fragments.
// Fragment layout per (m, kk, nt): 32 lanes x 16 elems; lane L elem e holds
// W[kk*32 + (e>>3)*16 + (L>>4)*8 + (e&7), nt*16 + (L&15)].  196608 total elements.
// ---------------------------------------------------------------------------
__global__ void wfrag_prep(const float* __restrict__ Wq,
                           const float* __restrict__ Wk,
                           const float* __restrict__ Wv,
                           __bf16* __restrict__ wfrag) {
  int idx  = blockIdx.x * 256 + threadIdx.x;   // grid sized exactly
  int e    = idx & 15;
  int lane = (idx >> 4) & 31;
  int nt   = (idx >> 9) & 3;
  int kk   = (idx >> 11) & 31;
  int m    = idx >> 16;                        // 0..2
  const float* W = (m == 0) ? Wq : (m == 1) ? Wk : Wv;
  int krow = kk * 32 + ((e >> 3) << 4) + ((lane >> 4) << 3) + (e & 7);
  int col  = nt * 16 + (lane & 15);
  wfrag[idx] = (__bf16)W[krow * HDIM + col];
}

// ---------------------------------------------------------------------------
// Kernel 1: QKV projection. 8 waves/block, one 16-row tile each, marching the
// 32 K-steps in lockstep. The 12 KB of W fragments per K-step are shared by
// all 8 waves, so they are staged into LDS (double-buffered) with the CDNA5
// async copy path (global_load_async_to_lds_b128 / ASYNCcnt) overlapped with
// the 12 WMMAs of the current K-step. B fragments are preloaded in groups of
// 4 into distinct registers so the DS waits stagger instead of fencing every
// WMMA.
// Q,K stored bf16 row-major [B*T, 64]; V stored transposed bf16 [B, 64, T].
// ---------------------------------------------------------------------------
__global__ void qkv_proj(const float* __restrict__ x,
                         const __bf16* __restrict__ wfrag,
                         __bf16* __restrict__ Qb,
                         __bf16* __restrict__ Kb,
                         __bf16* __restrict__ Vt) {
  // 2 buffers x 12 fragments x 512 bf16 (1 KB each) = 24 KB LDS.
  __shared__ __attribute__((aligned(64))) __bf16 wstage[2][12 * 512];

  const int tid  = threadIdx.x;
  const int wave = tid >> 5;
  const int lane = tid & 31;
  const int half = lane >> 4;
  const int r    = lane & 15;
  const int rowbase = (blockIdx.x * 8 + wave) * 16;

  // Copy the 12 fragments (12288 B) of K-step kk into wstage[buf].
  // 768 chunks of 16 B; each of the 256 threads moves 3 chunks, async into LDS.
  auto issue_copy = [&](int kk, int buf) {
#pragma unroll
    for (int i = 0; i < 3; ++i) {
      const int c  = tid + 256 * i;    // chunk id 0..767
      const int f  = c >> 6;           // fragment 0..11 (64 x 16B chunks each)
      const int m  = f >> 2;
      const int nt = f & 3;
      const __bf16* g = wfrag + (size_t)((m * 32 + kk) * 4 + nt) * 512 + (c & 63) * 8;
      __bf16* l = &wstage[buf][f * 512 + (c & 63) * 8];
      unsigned lds_off = (unsigned)(size_t)l;            // LDS aperture: addr[31:0]
      unsigned long long ga = (unsigned long long)(size_t)g;
      asm volatile("global_load_async_to_lds_b128 %0, %1, off"
                   :: "v"(lds_off), "v"(ga) : "memory");
    }
  };

  v8f acc[3][4] = {};
  const float* xrow = x + (size_t)(rowbase + r) * CDIM;

  issue_copy(0, 0);
  int cur = 0;
  for (int kk = 0; kk < 32; ++kk) {
    // Make this wave's async LDS writes complete, then make them visible
    // block-wide (and guarantee everyone is done reading the other buffer).
    asm volatile("s_wait_asynccnt 0" ::: "memory");
    __syncthreads();
    if (kk + 1 < 32) issue_copy(kk + 1, cur ^ 1);

    // A fragment of X (16x32 bf16): lane holds row (lane&15),
    // cols {half*8 + 0..7} and {16 + half*8 + 0..7}.
    const float* xp = xrow + kk * 32 + half * 8;
    float4 c00 = *(const float4*)(xp);
    float4 c01 = *(const float4*)(xp + 4);
    float4 c10 = *(const float4*)(xp + 16);
    float4 c11 = *(const float4*)(xp + 20);
    v16bf a;
    a[0]  = (__bf16)c00.x; a[1]  = (__bf16)c00.y; a[2]  = (__bf16)c00.z; a[3]  = (__bf16)c00.w;
    a[4]  = (__bf16)c01.x; a[5]  = (__bf16)c01.y; a[6]  = (__bf16)c01.z; a[7]  = (__bf16)c01.w;
    a[8]  = (__bf16)c10.x; a[9]  = (__bf16)c10.y; a[10] = (__bf16)c10.z; a[11] = (__bf16)c10.w;
    a[12] = (__bf16)c11.x; a[13] = (__bf16)c11.y; a[14] = (__bf16)c11.z; a[15] = (__bf16)c11.w;

#pragma unroll
    for (int m = 0; m < 3; ++m) {
      v16bf bfr[4];
#pragma unroll
      for (int nt = 0; nt < 4; ++nt)
        bfr[nt] = *(const v16bf*)&wstage[cur][(m * 4 + nt) * 512 + lane * 16];
#pragma unroll
      for (int nt = 0; nt < 4; ++nt)
        acc[m][nt] = wmma_bf16(a, bfr[nt], acc[m][nt]);
    }
    cur ^= 1;
  }

  // D layout: lane L, vgpr v -> row = v + (L>>4)*8, col = L&15.
  const int bidx = rowbase / TSEQ;
  const int trow = rowbase % TSEQ;
#pragma unroll
  for (int nt = 0; nt < 4; ++nt) {
#pragma unroll
    for (int v = 0; v < 8; ++v) {
      int row = rowbase + v + half * 8;
      int col = nt * 16 + r;
      Qb[(size_t)row * HDIM + col] = (__bf16)acc[0][nt][v];
      Kb[(size_t)row * HDIM + col] = (__bf16)acc[1][nt][v];
      int t = trow + v + half * 8;
      Vt[((size_t)bidx * HDIM + col) * TSEQ + t] = (__bf16)acc[2][nt][v];
    }
  }
}

// ---------------------------------------------------------------------------
// Kernel 2: causal flash attention. One wave owns one 16-query tile,
// streams 32-key chunks with online softmax, accumulates O in fp32.
// K/V fragments for the next chunk are prefetched (global_prefetch_b8) while
// the current chunk's WMMAs and softmax execute.
// ---------------------------------------------------------------------------
__global__ void attn(const __bf16* __restrict__ Qb,
                     const __bf16* __restrict__ Kb,
                     const __bf16* __restrict__ Vt,
                     float* __restrict__ out) {
  __shared__ __attribute__((aligned(64))) __bf16 pbuf[8][16 * 32];

  const int wave = threadIdx.x >> 5;
  const int lane = threadIdx.x & 31;
  const int half = lane >> 4;
  const int r    = lane & 15;
  const int g    = blockIdx.x * 8 + wave;      // 0..1023 query tiles
  const int b    = g >> 7;                     // 128 tiles per batch
  const int qbase = (g & 127) * 16;

  // Q A-fragments for hh = 0,1 (H split into two K=32 slabs).
  v16bf qf[2];
#pragma unroll
  for (int hh = 0; hh < 2; ++hh) {
    const __bf16* qp = Qb + (size_t)(b * TSEQ + qbase + r) * HDIM + hh * 32 + half * 16;
    v8bf c0 = *(const v8bf*)(qp);
    v8bf c1 = *(const v8bf*)(qp + 8);
    qf[hh] = __builtin_shufflevector(c0, c1, 0, 1, 2, 3, 4, 5, 6, 7,
                                     8, 9, 10, 11, 12, 13, 14, 15);
  }

  float m_i[8], l_i[8];
  v8f acc[4] = {};
#pragma unroll
  for (int v = 0; v < 8; ++v) { m_i[v] = -INFINITY; l_i[v] = 0.f; }

  const float scale = 0.125f;  // 64^-0.5
  const int nkb = (qbase + 16 + 31) >> 5;      // 32-key chunks covering [0, qbase+16)

  for (int jc = 0; jc < nkb; ++jc) {
    const int kb = jc * 32;

    // Prefetch next chunk's K and V lines behind this chunk's compute.
    if (jc + 1 < nkb) {
      const int kn = kb + 32;
#pragma unroll
      for (int t2 = 0; t2 < 2; ++t2)
#pragma unroll
        for (int hh = 0; hh < 2; ++hh)
          __builtin_prefetch(Kb + (size_t)(b * TSEQ + kn + t2 * 16 + r) * HDIM
                                + hh * 32 + half * 16, 0, 3);
#pragma unroll
      for (int nt = 0; nt < 4; ++nt)
        __builtin_prefetch(Vt + (size_t)(b * HDIM + nt * 16 + r) * TSEQ
                              + kn + half * 16, 0, 3);
    }

    // Two 16x16 score tiles (keys kb..kb+15 and kb+16..kb+31).
    // Load all four K B-fragments into distinct registers first so the
    // load waits stagger instead of fencing each WMMA.
    v16bf kfr[4];
#pragma unroll
    for (int t2 = 0; t2 < 2; ++t2)
#pragma unroll
      for (int hh = 0; hh < 2; ++hh)
        kfr[t2 * 2 + hh] = *(const v16bf*)(Kb + (size_t)(b * TSEQ + kb + t2 * 16 + r) * HDIM
                                              + hh * 32 + half * 16);
    v8f s[2] = {};
    s[0] = wmma_bf16(qf[0], kfr[0], s[0]);
    s[0] = wmma_bf16(qf[1], kfr[1], s[0]);
    s[1] = wmma_bf16(qf[0], kfr[2], s[1]);
    s[1] = wmma_bf16(qf[1], kfr[3], s[1]);

    const bool need_mask = (kb + 31 > qbase);
#pragma unroll
    for (int v = 0; v < 8; ++v) {
      const int row = qbase + v + half * 8;
      float s0 = s[0][v] * scale;
      float s1 = s[1][v] * scale;
      if (need_mask) {
        if (kb + r > row)      s0 = -INFINITY;
        if (kb + 16 + r > row) s1 = -INFINITY;
      }
      float rm = hmax16(fmaxf(s0, s1));
      float nm = fmaxf(m_i[v], rm);
      float p0 = __expf(s0 - nm);
      float p1 = __expf(s1 - nm);
      float rs = hsum16(p0 + p1);
      float alpha = __expf(m_i[v] - nm);
      l_i[v] = l_i[v] * alpha + rs;
      m_i[v] = nm;
#pragma unroll
      for (int nt = 0; nt < 4; ++nt) acc[nt][v] *= alpha;
      // Stage probs row-major 16x32 in LDS for D-layout -> A-layout conversion.
      pbuf[wave][(v + half * 8) * 32 + r]      = (__bf16)p0;
      pbuf[wave][(v + half * 8) * 32 + 16 + r] = (__bf16)p1;
    }

    // Wave-private LDS write->read fence (waves diverge in trip count,
    // so no block barrier; DS ops from one wave complete in order).
    __asm__ volatile("s_wait_dscnt 0" ::: "memory");

    // A fragment of P (16x32): row = lane&15, cols {half*16+e}, e = 0..15.
    const __bf16* pw = &pbuf[wave][0];
    v8bf pc0 = *(const v8bf*)(pw + r * 32 + half * 16);
    v8bf pc1 = *(const v8bf*)(pw + r * 32 + half * 16 + 8);
    v16bf pa = __builtin_shufflevector(pc0, pc1, 0, 1, 2, 3, 4, 5, 6, 7,
                                       8, 9, 10, 11, 12, 13, 14, 15);

    // O += P(16x32) @ V(32x16) per 16-wide h tile; Vt is [b][h][t] so the
    // B fragment (k = key index) is a contiguous 32B/lane load.
    v16bf vfr[4];
#pragma unroll
    for (int nt = 0; nt < 4; ++nt)
      vfr[nt] = *(const v16bf*)(Vt + (size_t)(b * HDIM + nt * 16 + r) * TSEQ
                                   + kb + half * 16);
#pragma unroll
    for (int nt = 0; nt < 4; ++nt)
      acc[nt] = wmma_bf16(pa, vfr[nt], acc[nt]);
  }

  // Epilogue: normalize by row sums, fp32 out [B, T, 64].
#pragma unroll
  for (int nt = 0; nt < 4; ++nt) {
#pragma unroll
    for (int v = 0; v < 8; ++v) {
      const int row = qbase + v + half * 8;
      out[(size_t)(b * TSEQ + row) * HDIM + nt * 16 + r] = acc[nt][v] / l_i[v];
    }
  }
}

// ---------------------------------------------------------------------------
// Workspace layout (bytes):
//   [0,   2MB) Qb  bf16 [B*T, 64]
//   [2MB, 4MB) Kb  bf16 [B*T, 64]
//   [4MB, 6MB) Vt  bf16 [B, 64, T]
//   [6MB, 6MB+384KB) W fragments bf16
// ---------------------------------------------------------------------------
extern "C" void kernel_launch(void* const* d_in, const int* in_sizes, int n_in,
                              void* d_out, int out_size, void* d_ws, size_t ws_size,
                              hipStream_t stream) {
  (void)in_sizes; (void)n_in; (void)out_size; (void)ws_size;
  const float* x  = (const float*)d_in[0];
  const float* Wq = (const float*)d_in[1];
  const float* Wk = (const float*)d_in[2];
  const float* Wv = (const float*)d_in[3];
  float* out = (float*)d_out;

  char* ws = (char*)d_ws;
  __bf16* Qb = (__bf16*)(ws + (size_t)0);
  __bf16* Kb = (__bf16*)(ws + ((size_t)2 << 20));
  __bf16* Vt = (__bf16*)(ws + ((size_t)4 << 20));
  __bf16* Wf = (__bf16*)(ws + ((size_t)6 << 20));

  wfrag_prep<<<768, 256, 0, stream>>>(Wq, Wk, Wv, Wf);        // 196608 threads
  qkv_proj<<<128, 256, 0, stream>>>(x, Wf, Qb, Kb, Vt);       // 1024 waves
  attn<<<128, 256, 0, stream>>>(Qb, Kb, Vt, out);             // 1024 waves
}
